// NodeAttributeAggregator_32031866093969
// MI455X (gfx1250) — matrix operand order repaired and checked by reading, not compiled, not run on hardware
//
#include <hip/hip_runtime.h>
#include <hip/hip_bf16.h>
#include <stddef.h>

// ---------------------------------------------------------------------------
// GCN on MI455X (gfx1250, wave32, WMMA).
//  - 4 GEMMs (50000x256 @ 256x256) via V_WMMA_F32_16X16X32_BF16
//    one wave per 16x128 output tile: 8 accumulators, 8 WMMAs per K-step,
//    A fragment (f32->bf16 + optional fused ReLU) amortized over 8 WMMAs.
//  - symmetric-norm aggregation via per-edge wave scatter with f32 atomics
// Working set (~65MB) fits in the 192MB L2, so aggregation stays on-chip.
// ---------------------------------------------------------------------------

typedef __attribute__((ext_vector_type(16))) __bf16 v16bf;
typedef __attribute__((ext_vector_type(8)))  float  v8f;

#define N_NODES 50000
#define N_EDGES 800000
#define DH      256

// ---- weight transpose + f32->bf16 convert: Wt[n*256 + k] = W[k*256 + n] ----
__global__ void wt_bf16_kernel(const float* __restrict__ W,
                               __bf16* __restrict__ Wt) {
    int idx = blockIdx.x * blockDim.x + threadIdx.x;   // 65536 threads
    int k = idx & (DH - 1);
    int n = idx >> 8;
    Wt[(size_t)n * DH + k] = (__bf16)W[(size_t)k * DH + n];
}

// ---- degree (with self loop) -> dinv = rsqrt(deg) ----
__global__ void deg_init_kernel(float* __restrict__ deg) {
    int i = blockIdx.x * blockDim.x + threadIdx.x;
    if (i < N_NODES) deg[i] = 1.0f;                    // self loop
}
__global__ void deg_edges_kernel(const long long* __restrict__ dst,
                                 float* __restrict__ deg) {
    int i = blockIdx.x * blockDim.x + threadIdx.x;
    if (i < N_EDGES) atomicAdd(&deg[dst[i]], 1.0f);
}
__global__ void dinv_kernel(float* __restrict__ deg) {
    int i = blockIdx.x * blockDim.x + threadIdx.x;
    if (i < N_NODES) {
        float d = deg[i];
        deg[i] = (d > 0.0f) ? rsqrtf(d) : 0.0f;
    }
}

// ---- GEMM: C[M x 256] = (relu_in? relu(A) : A) @ W  (+ bias) --------------
// A is f32 row-major [M x 256]; Wt is bf16 "transposed" [n*256 + k].
// One wave (32 threads) computes a 16x128 output tile: 8 accumulators,
// 8 K-steps of 8x v_wmma_f32_16x16x32_bf16.  No branches -> EXEC all ones.
__global__ void __launch_bounds__(32)
gemm_wmma_kernel(const float* __restrict__ A,
                 const __bf16* __restrict__ Wt,
                 const float* __restrict__ bias,
                 float* __restrict__ C,
                 int use_bias, int relu_in) {
    const int lane = threadIdx.x;          // 0..31
    const int lm   = lane & 15;
    const int half = lane >> 4;            // 0 | 1
    const int r0   = blockIdx.x * 16;      // row tile (M)
    const int c0   = blockIdx.y * 128;     // col tile (N)

    const float4* Arow = (const float4*)(A + (size_t)(r0 + lm) * DH);

    v8f acc[8] = {v8f{0}, v8f{0}, v8f{0}, v8f{0},
                  v8f{0}, v8f{0}, v8f{0}, v8f{0}};

    for (int kk = 0; kk < DH; kk += 32) {
        // A fragment, 16-bit A 16x32 layout (ISA 7.12.2):
        // a[g*8 + j] = A[row, kk + g*16 + half*8 + j]  -> two b128 loads per g
        v16bf a;
        #pragma unroll
        for (int g = 0; g < 2; ++g) {
            int kbase = kk + g * 16 + half * 8;        // multiple of 8 floats
            float4 f0 = Arow[(kbase >> 2) + 0];
            float4 f1 = Arow[(kbase >> 2) + 1];
            float v[8] = {f0.x, f0.y, f0.z, f0.w, f1.x, f1.y, f1.z, f1.w};
            #pragma unroll
            for (int j = 0; j < 8; ++j) {
                float f = v[j];
                if (relu_in) f = fmaxf(f, 0.0f);       // v_max_f32
                a[g * 8 + j] = (__bf16)f;
            }
        }
        // B fragments, 16-bit B 32x16 layout: b[g*8 + j] at
        // k = kk + g*16 + half*8 + j (two b128 loads per fragment).
        #pragma unroll
        for (int t = 0; t < 8; ++t) {
            const __bf16* Bcol =
                Wt + (size_t)(c0 + t * 16 + lm) * DH + kk + half * 8;
            v16bf b;
            #pragma unroll
            for (int i = 0; i < 16; ++i) b[i] = Bcol[(i >> 3) * 16 + (i & 7)];
            acc[t] = __builtin_amdgcn_wmma_f32_16x16x32_bf16(
                false, a, false, b, (short)0, acc[t], false, false);
        }
    }

    // C/D layout: VGPR m, lanes 0-15 -> M=m ; lanes 16-31 -> M=m+8 ; N=lane&15
    #pragma unroll
    for (int t = 0; t < 8; ++t) {
        int col = c0 + t * 16 + lm;
        float bv = use_bias ? bias[col] : 0.0f;
        #pragma unroll
        for (int m = 0; m < 8; ++m) {
            C[(size_t)(r0 + half * 8 + m) * DH + col] = acc[t][m] + bv;
        }
    }
}

// ---- aggregation: out = selfloop + bias, then edge scatter ----------------
__global__ void agg_init_kernel(const float* __restrict__ t,
                                const float* __restrict__ dinv,
                                const float* __restrict__ bias,
                                float* __restrict__ out) {
    size_t idx = (size_t)blockIdx.x * 256 + threadIdx.x;   // N_NODES*256
    int node = (int)(idx >> 8);
    int col  = (int)(idx & (DH - 1));
    float di = dinv[node];
    out[idx] = t[idx] * di * di + bias[col];
}

// one wave per edge: lanes stride the 256 columns (coalesced 128B loads)
__global__ void agg_edges_kernel(const float* __restrict__ t,
                                 const float* __restrict__ dinv,
                                 const long long* __restrict__ src,
                                 const long long* __restrict__ dst,
                                 float* __restrict__ out) {
    int edge = blockIdx.x * 8 + (threadIdx.x >> 5);
    if (edge >= N_EDGES) return;
    int lane = threadIdx.x & 31;
    long long s = src[edge];
    long long d = dst[edge];
    float nrm = dinv[s] * dinv[d];
    const float* ts = t + (size_t)s * DH;
    float* od = out + (size_t)d * DH;
    #pragma unroll
    for (int i = 0; i < 8; ++i) {
        int c = lane + i * 32;
        atomicAdd(od + c, ts[c] * nrm);
    }
}

// ---------------------------------------------------------------------------
extern "C" void kernel_launch(void* const* d_in, const int* in_sizes, int n_in,
                              void* d_out, int out_size, void* d_ws, size_t ws_size,
                              hipStream_t stream) {
    const float*     x     = (const float*)d_in[0];
    const long long* ei    = (const long long*)d_in[1];  // [2, E] int64
    const float*     Wpre  = (const float*)d_in[2];
    const float*     bpre  = (const float*)d_in[3];
    const float*     W1    = (const float*)d_in[4];
    const float*     b1    = (const float*)d_in[5];
    const float*     W2    = (const float*)d_in[6];
    const float*     b2    = (const float*)d_in[7];
    const float*     Wpost = (const float*)d_in[8];
    const float*     bpost = (const float*)d_in[9];
    float*           y     = (float*)d_out;

    const long long* src = ei;
    const long long* dst = ei + N_EDGES;

    // workspace layout
    char*  ws     = (char*)d_ws;
    size_t hbytes = (size_t)N_NODES * DH * sizeof(float);       // 51.2 MB
    float*  hA     = (float*)(ws);
    float*  hB     = (float*)(ws + hbytes);
    float*  hC     = (float*)(ws + 2 * hbytes);
    __bf16* WtPre  = (__bf16*)(ws + 3 * hbytes);
    __bf16* Wt1    = WtPre  + (size_t)DH * DH;
    __bf16* Wt2    = Wt1    + (size_t)DH * DH;
    __bf16* WtPost = Wt2    + (size_t)DH * DH;
    float*  dinv   = (float*)(WtPost + (size_t)DH * DH);

    // convert weights (bf16, transposed so B-fragments are K-contiguous)
    int wthreads = DH * DH;
    wt_bf16_kernel<<<wthreads / 256, 256, 0, stream>>>(Wpre,  WtPre);
    wt_bf16_kernel<<<wthreads / 256, 256, 0, stream>>>(W1,    Wt1);
    wt_bf16_kernel<<<wthreads / 256, 256, 0, stream>>>(W2,    Wt2);
    wt_bf16_kernel<<<wthreads / 256, 256, 0, stream>>>(Wpost, WtPost);

    // degrees -> dinv
    deg_init_kernel <<<(N_NODES + 255) / 256, 256, 0, stream>>>(dinv);
    deg_edges_kernel<<<(N_EDGES + 255) / 256, 256, 0, stream>>>(dst, dinv);
    dinv_kernel     <<<(N_NODES + 255) / 256, 256, 0, stream>>>(dinv);

    dim3 ggrid(N_NODES / 16, DH / 128);  // 3125 x 2 wave-tiles (16x128 each)
    dim3 gblock(32);                     // one wave32 per tile

    // h0 = x @ Wpre + bpre
    gemm_wmma_kernel<<<ggrid, gblock, 0, stream>>>(x, WtPre, bpre, hA, 1, 0);

    // layer 1: t = h0 @ W1 ; agg (+b1)
    gemm_wmma_kernel<<<ggrid, gblock, 0, stream>>>(hA, Wt1, bpre, hB, 0, 0);
    agg_init_kernel <<<N_NODES, 256, 0, stream>>>(hB, dinv, b1, hC);
    agg_edges_kernel<<<N_EDGES / 8, 256, 0, stream>>>(hB, dinv, src, dst, hC);

    // layer 2: t = relu(agg1) @ W2 ; agg (+b2)
    gemm_wmma_kernel<<<ggrid, gblock, 0, stream>>>(hC, Wt2, bpre, hA, 0, 1);
    agg_init_kernel <<<N_NODES, 256, 0, stream>>>(hA, dinv, b2, hB);
    agg_edges_kernel<<<N_EDGES / 8, 256, 0, stream>>>(hA, dinv, src, dst, hB);

    // y = relu(agg2) @ Wpost + bpost
    gemm_wmma_kernel<<<ggrid, gblock, 0, stream>>>(hB, WtPost, bpost, y, 1, 1);
}
